// InvestigationBlock_8693013807150
// MI455X (gfx1250) — compile-verified, matrix operand order checked by reading, not run on hardware
//
#include <hip/hip_runtime.h>

typedef __attribute__((ext_vector_type(16))) _Float16 v16h;
typedef __attribute__((ext_vector_type(8)))  _Float16 v8h;
typedef __attribute__((ext_vector_type(8)))  float    v8f;

#define DIM      768
#define HEADS    12
#define HEAD_DIM 64
#define HIDDEN   3072
#define SEQ      2048
#define BATCH    4
#define ROWS     (BATCH * SEQ)          // 8192
#define ATT_SCALE 0.125f                // 64^-0.5

union FragU { v16h v; v8h h[2]; };

// Load a 16x32 f16 A/B fragment from an LDS tile (row-major, `ld` halves per row).
// ISA 7.12.2: lanes 0-15 hold row=lane, K={0..7,16..23}; lanes 16-31 row=lane-16,
// K={8..15,24..31}.  B fragments use the same pattern on an N x K staged tile.
__device__ inline v16h load_frag16(const _Float16* base, int ld) {
  const int lane = threadIdx.x & 31;
  const int r    = lane & 15;
  const int ko   = (lane >> 4) << 3;
  const _Float16* p = base + r * ld + ko;
  FragU f;
  f.h[0] = *(const v8h*)(p);
  f.h[1] = *(const v8h*)(p + 16);
  return f.v;
}

__device__ inline v8f wmma_f16(v16h a, v16h b, v8f c) {
  return __builtin_amdgcn_wmma_f32_16x16x32_f16(false, a, false, b, (short)0, c,
                                                false, false);
}

// gfx1250 async global->LDS copy (ASYNCcnt-tracked, VGPR-bypassing).
// VDST = LDS byte offset (low 32 bits of a generic shared pointer = AS(3) offset),
// VADDR = 64-bit global address, SADDR = off.
__device__ inline void async_copy_b128(void* lds_ptr, const void* gptr) {
  unsigned int dst = (unsigned int)(unsigned long long)lds_ptr;
  asm volatile("global_load_async_to_lds_b128 %0, %1, off"
               :: "v"(dst), "v"(gptr) : "memory");
}
__device__ inline void async_wait0() {
  asm volatile("s_wait_asynccnt 0x0" ::: "memory");
}

// ---------------------------------------------------------------- converts
__global__ __launch_bounds__(256) void cvt_f32_f16(const float* __restrict__ in,
                                                   _Float16* __restrict__ out, int n) {
  int i = blockIdx.x * 256 + threadIdx.x;
  if (i < n) out[i] = (_Float16)in[i];
}

// ---------------------------------------------------------------- LayerNorm
__global__ __launch_bounds__(256)
void ln_kernel(const float* __restrict__ x, const float* __restrict__ g,
               const float* __restrict__ be, _Float16* __restrict__ out) {
  const int row = blockIdx.x;
  const int t = threadIdx.x;
  const float* xr = x + (size_t)row * DIM;
  float v0 = xr[t], v1 = xr[t + 256], v2 = xr[t + 512];
  float s  = v0 + v1 + v2;
  float sq = v0 * v0 + v1 * v1 + v2 * v2;
#pragma unroll
  for (int o = 16; o; o >>= 1) { s += __shfl_xor(s, o, 32); sq += __shfl_xor(sq, o, 32); }
  __shared__ float rs[8], rq[8];
  if ((t & 31) == 0) { rs[t >> 5] = s; rq[t >> 5] = sq; }
  __syncthreads();
  float ts = 0.f, tq = 0.f;
#pragma unroll
  for (int i = 0; i < 8; ++i) { ts += rs[i]; tq += rq[i]; }
  const float mu   = ts * (1.0f / DIM);
  const float var  = tq * (1.0f / DIM) - mu * mu;
  const float rinv = rsqrtf(var + 1e-5f);
  _Float16* o = out + (size_t)row * DIM;
  o[t]       = (_Float16)((v0 - mu) * rinv * g[t]       + be[t]);
  o[t + 256] = (_Float16)((v1 - mu) * rinv * g[t + 256] + be[t + 256]);
  o[t + 512] = (_Float16)((v2 - mu) * rinv * g[t + 512] + be[t + 512]);
}

// ---------------------------------------------------------------- WMMA GEMM
// C[128x64] tile per 128-thread block (4 waves; wave w -> rows w*32..w*32+31,
// i.e. 2 M-frags x 4 N-frags = 8 WMMA per K-step).  A tile is staged with
// async global->LDS copies and double-buffered; B is loaded coalesced along N
// and scattered transposed (N x K) into LDS.
enum { EPI_QKV = 0, EPI_RES = 1, EPI_GELU = 2 };

template <int EPI>
__global__ __launch_bounds__(128)
void gemm128(const _Float16* __restrict__ A, const _Float16* __restrict__ Bw,
             const float* __restrict__ bias,
             const float* __restrict__ res, float* __restrict__ outf,
             _Float16* __restrict__ outh,
             _Float16* __restrict__ qo, _Float16* __restrict__ ko,
             _Float16* __restrict__ vo,
             const float* __restrict__ ga, const float* __restrict__ gb,
             const float* __restrict__ gc,
             int M, int N, int K) {
  __shared__ alignas(16) _Float16 As[2][128][40];  // 128 x 32, pitch 40 (16B rows)
  __shared__ alignas(16) _Float16 Bs[2][64][40];   // N x K staged (transposed)
  const int t  = threadIdx.x;
  const int wv = t >> 5;
  const int m0 = blockIdx.y * 128;
  const int n0 = blockIdx.x * 64;

  auto load_tiles = [&](int buf, int k0) {
    // A tile: 128 rows x 32 halves = 512 b128 chunks, async to LDS
    for (int c = t; c < 512; c += 128) {
      int row = c >> 2, seg = c & 3;
      async_copy_b128(&As[buf][row][seg * 8],
                      A + (size_t)(m0 + row) * K + k0 + seg * 8);
    }
    // B tile: 32 K-rows coalesced along N, scatter transposed into Bs[N][K]
    for (int c = t; c < 256; c += 128) {
      int kk = c >> 3, ch = c & 7;
      v8h bv = *(const v8h*)(Bw + (size_t)(k0 + kk) * N + n0 + ch * 8);
#pragma unroll
      for (int j = 0; j < 8; ++j) Bs[buf][ch * 8 + j][kk] = bv[j];
    }
  };

  v8f acc[2][4] = {};
  const int nsteps = K / 32;
  int buf = 0;
  load_tiles(0, 0);
  for (int kt = 0; kt < nsteps; ++kt) {
    async_wait0();       // this wave's async tile copies complete
    __syncthreads();     // all waves' copies + transposes visible
    if (kt + 1 < nsteps) load_tiles(buf ^ 1, (kt + 1) * 32);  // prefetch next
    v16h a0 = load_frag16(&As[buf][wv * 32][0], 40);
    v16h a1 = load_frag16(&As[buf][wv * 32 + 16][0], 40);
#pragma unroll
    for (int nt = 0; nt < 4; ++nt) {
      v16h b = load_frag16(&Bs[buf][nt * 16][0], 40);
      acc[0][nt] = wmma_f16(a0, b, acc[0][nt]);
      acc[1][nt] = wmma_f16(a1, b, acc[1][nt]);
    }
    buf ^= 1;
  }

  // epilogue — C layout: VGPR r -> row rbase+r, col = lane&15 (+16*ntile)
  const int lane  = t & 31;
  const int colid = lane & 15;
  const int rbase = (lane >> 4) << 3;
  float ca = 0.f, cb = 0.f, cc = 0.f;
  if constexpr (EPI == EPI_GELU) { ca = *ga; cb = *gb; cc = *gc; }
#pragma unroll
  for (int mt = 0; mt < 2; ++mt) {
#pragma unroll
    for (int nt = 0; nt < 4; ++nt) {
#pragma unroll
      for (int r = 0; r < 8; ++r) {
        const int row = m0 + wv * 32 + mt * 16 + rbase + r;
        const int col = n0 + nt * 16 + colid;
        float val = acc[mt][nt][r] + bias[col];
        if constexpr (EPI == EPI_GELU) {
          val = ca * val * val + cb * val + cc;
          outh[(size_t)row * N + col] = (_Float16)val;
        } else if constexpr (EPI == EPI_RES) {
          const size_t idx = (size_t)row * N + col;
          outf[idx] = res[idx] + val;
        } else {  // EPI_QKV: scatter bias-applied qkv into head-major [B,H,N,D]
          const int which = col / DIM;
          const int rem = col - which * DIM;
          const int h = rem >> 6, d = rem & 63;
          const int b = row >> 11, n = row & 2047;
          const size_t dst = (((size_t)(b * HEADS + h)) * SEQ + n) * HEAD_DIM + d;
          const _Float16 hv = (_Float16)val;
          if (which == 0) qo[dst] = hv;
          else if (which == 1) ko[dst] = hv;
          else vo[dst] = hv;
        }
      }
    }
  }
}

// ---------------------------------------------------------------- attention
// Single-pass poly-normalized attention: per (b,h) and 64-query tile, stream
// 32-key chunks; S = q k^T via WMMA, w = clamp(poly(S*scale)), accumulate
// rowsum and O += w V (second WMMA, w bounced through LDS for A-layout).
__global__ __launch_bounds__(128)
void attn_kernel(const _Float16* __restrict__ q, const _Float16* __restrict__ k,
                 const _Float16* __restrict__ v, _Float16* __restrict__ out,
                 const float* __restrict__ pa, const float* __restrict__ pb,
                 const float* __restrict__ pc) {
  __shared__ alignas(16) _Float16 Qs[64][72];      // 64 q-rows x 64 dims
  __shared__ alignas(16) _Float16 Ks[32][72];      // 32 keys  x 64 dims (B: N x K)
  __shared__ alignas(16) _Float16 Vt[64][40];      // dims x keys (B: N x K for w@V)
  __shared__ alignas(16) _Float16 Ws[4][16][40];   // per-wave w tile 16 x 32
  const int t    = threadIdx.x;
  const int wv   = t >> 5;
  const int lane = t & 31;
  const int bh = blockIdx.y;
  const int b = bh / HEADS, h = bh - b * HEADS;
  const int q0 = blockIdx.x * 64;
  const float ca = *pa, cb = *pb, cc = *pc;
  const size_t base = (size_t)bh * SEQ * HEAD_DIM;

  for (int c = t; c < 512; c += 128) {   // Q tile: 64 x 64 halves
    int row = c >> 3, seg = c & 7;
    *(v8h*)&Qs[row][seg * 8] =
        *(const v8h*)(q + base + (size_t)(q0 + row) * HEAD_DIM + seg * 8);
  }

  v8f O[4] = {};
  float rs[8] = {0.f, 0.f, 0.f, 0.f, 0.f, 0.f, 0.f, 0.f};
  const int colid = lane & 15;
  const int rbase = (lane >> 4) << 3;

  for (int m0 = 0; m0 < SEQ; m0 += 32) {
    __syncthreads();
    for (int c = t; c < 256; c += 128) {   // K chunk: 32 x 64
      int row = c >> 3, seg = c & 7;
      *(v8h*)&Ks[row][seg * 8] =
          *(const v8h*)(k + base + (size_t)(m0 + row) * HEAD_DIM + seg * 8);
    }
    for (int c = t; c < 2048; c += 128) {  // V chunk transposed: Vt[d][m]
      int m = c >> 6, d = c & 63;
      Vt[d][m] = v[base + (size_t)(m0 + m) * HEAD_DIM + d];
    }
    __syncthreads();

    v8f s0 = {}, s1 = {};
#pragma unroll
    for (int ks = 0; ks < 2; ++ks) {       // head dim 64 -> two x32 K-steps
      v16h a  = load_frag16(&Qs[wv * 16][ks * 32], 72);
      v16h b0 = load_frag16(&Ks[0][ks * 32], 72);
      v16h b1 = load_frag16(&Ks[16][ks * 32], 72);
      s0 = wmma_f16(a, b0, s0);
      s1 = wmma_f16(a, b1, s1);
    }
#pragma unroll
    for (int r = 0; r < 8; ++r) {
      float x0 = s0[r] * ATT_SCALE; x0 = ca * x0 * x0 + cb * x0 + cc; x0 = fmaxf(x0, 1e-6f);
      float x1 = s1[r] * ATT_SCALE; x1 = ca * x1 * x1 + cb * x1 + cc; x1 = fmaxf(x1, 1e-6f);
      rs[r] += x0 + x1;
      Ws[wv][rbase + r][colid]      = (_Float16)x0;
      Ws[wv][rbase + r][16 + colid] = (_Float16)x1;
    }
    v16h aw = load_frag16(&Ws[wv][0][0], 40);
#pragma unroll
    for (int dt = 0; dt < 4; ++dt) {
      v16h bv = load_frag16(&Vt[dt * 16][0], 40);
      O[dt] = wmma_f16(aw, bv, O[dt]);
    }
  }

  // Row totals: reduce within each 16-lane group (matches C-layout row split).
#pragma unroll
  for (int r = 0; r < 8; ++r) {
    float s = rs[r];
    s += __shfl_xor(s, 1, 32);
    s += __shfl_xor(s, 2, 32);
    s += __shfl_xor(s, 4, 32);
    s += __shfl_xor(s, 8, 32);
    rs[r] = s + 1e-8f;
  }
#pragma unroll
  for (int dt = 0; dt < 4; ++dt) {
#pragma unroll
    for (int r = 0; r < 8; ++r) {
      const int row = q0 + wv * 16 + rbase + r;
      const float val = O[dt][r] / rs[r];
      out[((size_t)(b * SEQ + row)) * DIM + h * HEAD_DIM + dt * 16 + colid] =
          (_Float16)val;
    }
  }
}

// ---------------------------------------------------------------- launcher
extern "C" void kernel_launch(void* const* d_in, const int* in_sizes, int n_in,
                              void* d_out, int out_size, void* d_ws, size_t ws_size,
                              hipStream_t stream) {
  (void)in_sizes; (void)n_in; (void)out_size; (void)ws_size;
  const float* x      = (const float*)d_in[0];
  const float* ln1_g  = (const float*)d_in[1];
  const float* ln1_b  = (const float*)d_in[2];
  const float* ln2_g  = (const float*)d_in[3];
  const float* ln2_b  = (const float*)d_in[4];
  const float* qkv_w  = (const float*)d_in[5];
  const float* qkv_b  = (const float*)d_in[6];
  const float* proj_w = (const float*)d_in[7];
  const float* proj_b = (const float*)d_in[8];
  const float* fc1_w  = (const float*)d_in[9];
  const float* fc1_b  = (const float*)d_in[10];
  const float* fc2_w  = (const float*)d_in[11];
  const float* fc2_b  = (const float*)d_in[12];
  const float* attn_a = (const float*)d_in[13];
  const float* attn_b = (const float*)d_in[14];
  const float* attn_c = (const float*)d_in[15];
  const float* gelu_a = (const float*)d_in[16];
  const float* gelu_b = (const float*)d_in[17];
  const float* gelu_c = (const float*)d_in[18];
  float* outf = (float*)d_out;

  char* ws = (char*)d_ws;
  size_t off = 0;
  auto alloc_h = [&](size_t n_halves) {
    void* p = ws + off;
    off = (off + n_halves * sizeof(_Float16) + 255) & ~(size_t)255;
    return (_Float16*)p;
  };
  _Float16* wqkv = alloc_h((size_t)DIM * 3 * DIM);
  _Float16* wprj = alloc_h((size_t)DIM * DIM);
  _Float16* wfc1 = alloc_h((size_t)DIM * HIDDEN);
  _Float16* wfc2 = alloc_h((size_t)HIDDEN * DIM);
  _Float16* h1   = alloc_h((size_t)ROWS * DIM);
  _Float16* qb   = alloc_h((size_t)ROWS * DIM);
  _Float16* kb   = alloc_h((size_t)ROWS * DIM);
  _Float16* vb   = alloc_h((size_t)ROWS * DIM);
  _Float16* ao   = alloc_h((size_t)ROWS * DIM);
  _Float16* h2   = alloc_h((size_t)ROWS * DIM);
  _Float16* u    = alloc_h((size_t)ROWS * HIDDEN);

  // 1) weights f32 -> f16
  {
    int n;
    n = DIM * 3 * DIM;  cvt_f32_f16<<<(n + 255) / 256, 256, 0, stream>>>(qkv_w,  wqkv, n);
    n = DIM * DIM;      cvt_f32_f16<<<(n + 255) / 256, 256, 0, stream>>>(proj_w, wprj, n);
    n = DIM * HIDDEN;   cvt_f32_f16<<<(n + 255) / 256, 256, 0, stream>>>(fc1_w,  wfc1, n);
    n = HIDDEN * DIM;   cvt_f32_f16<<<(n + 255) / 256, 256, 0, stream>>>(fc2_w,  wfc2, n);
  }
  // 2) LN1
  ln_kernel<<<ROWS, 256, 0, stream>>>(x, ln1_g, ln1_b, h1);
  // 3) QKV GEMM (bias + head-major scatter epilogue)
  gemm128<EPI_QKV><<<dim3(3 * DIM / 64, ROWS / 128), 128, 0, stream>>>(
      h1, wqkv, qkv_b, nullptr, nullptr, nullptr, qb, kb, vb,
      nullptr, nullptr, nullptr, ROWS, 3 * DIM, DIM);
  // 4) streaming poly-normalized attention
  attn_kernel<<<dim3(SEQ / 64, BATCH * HEADS), 128, 0, stream>>>(
      qb, kb, vb, ao, attn_a, attn_b, attn_c);
  // 5) proj GEMM + residual(x) -> d_out holds x1 (fp32)
  gemm128<EPI_RES><<<dim3(DIM / 64, ROWS / 128), 128, 0, stream>>>(
      ao, wprj, proj_b, x, outf, nullptr, nullptr, nullptr, nullptr,
      nullptr, nullptr, nullptr, ROWS, DIM, DIM);
  // 6) LN2 on x1
  ln_kernel<<<ROWS, 256, 0, stream>>>(outf, ln2_g, ln2_b, h2);
  // 7) fc1 GEMM + poly-GELU -> u (f16)
  gemm128<EPI_GELU><<<dim3(HIDDEN / 64, ROWS / 128), 128, 0, stream>>>(
      h2, wfc1, fc1_b, nullptr, nullptr, u, nullptr, nullptr, nullptr,
      gelu_a, gelu_b, gelu_c, ROWS, HIDDEN, DIM);
  // 8) fc2 GEMM + residual(x1) -> final output (reads+writes own d_out element)
  gemm128<EPI_RES><<<dim3(DIM / 64, ROWS / 128), 128, 0, stream>>>(
      u, wfc2, fc2_b, outf, outf, nullptr, nullptr, nullptr, nullptr,
      nullptr, nullptr, nullptr, ROWS, DIM, HIDDEN);
}
